// GNNLLM_fsca_70188355551438
// MI455X (gfx1250) — matrix operand order repaired and checked by reading, not compiled, not run on hardware
//
#include <hip/hip_runtime.h>
#include <math.h>

// ---------------- model constants ----------------
#define BB 64
#define LLEN 1024
#define MFEAT 3
#define LEN_S 128
#define LEN_P 12
#define TT 140
#define DD 768
#define NHEAD 12
#define HD 64
#define NLAY 6
#define DMLP 3072
#define NTOK (BB * TT)          // 8960
#define E0 278                  // edges per batch: 2*(TT-1)
#define ZN (TT * DD)            // 107520

typedef __attribute__((ext_vector_type(16))) _Float16 v16h;
typedef __attribute__((ext_vector_type(8)))  float    v8f;

// LDS row padding (in halfs) to break bank conflicts: 144B stride
#define APAD 72
#define BPAD 72

// =====================================================================
// Generic WMMA GEMM:  C[M,N] = act(A[M,K] @ W[K,N] + bias) (+ resid)
// Block tile 128x64, K-stage 64 (2 wmma k-steps), 8 waves (wave32),
// each wave owns a 16x64 strip -> 8 v_wmma_f32_16x16x32_f16 per stage.
// A,W are f32 in memory, converted to f16 into padded LDS; f32 accum.
// FULL=true: M%128==0 && N%64==0 (no guards in the hot path).
// Requires K%64==0 (true for all uses: 768/3072/9216/98304).
// =====================================================================
template <bool FULL>
__global__ __launch_bounds__(256) void k_gemm(
    const float* __restrict__ A, int lda,
    const float* __restrict__ W, int ldw,
    const float* __restrict__ bias,
    const float* __restrict__ resid,
    float* __restrict__ C, int ldc,
    int Mdim, int Ndim, int Kdim, int act)
{
  __shared__ _Float16 As[128 * APAD];   // [row][k] k<64, padded
  __shared__ _Float16 Bs[64 * BPAD];    // transposed: [col][k] k<64, padded

  const int tid   = threadIdx.x;
  const int lane  = tid & 31;
  const int wave  = tid >> 5;
  const int khalf = lane >> 4;
  const int l15   = lane & 15;
  const int rowBase = blockIdx.y * 128;
  const int colBase = blockIdx.x * 64;

  v8f acc[4];
  v8f zero = {0.f, 0.f, 0.f, 0.f, 0.f, 0.f, 0.f, 0.f};
#pragma unroll
  for (int t = 0; t < 4; ++t) acc[t] = zero;

  for (int kb = 0; kb < Kdim; kb += 64) {
    // ---- cooperative A tile load: 128x64 f32 -> f16 LDS (coalesced float4)
#pragma unroll
    for (int q = 0; q < 8; ++q) {
      int idx = q * 256 + tid;           // 0..2047 float4 slots
      int row = idx >> 4;                // 16 float4 per row
      int c4  = (idx & 15) << 2;
      int r   = rowBase + row;
      float4 f = {0.f, 0.f, 0.f, 0.f};
      if (FULL || r < Mdim) {
        const float* src = A + (size_t)r * lda + kb + c4;
        f = *(const float4*)src;
        if (q == 0 && kb + 64 < Kdim) __builtin_prefetch(src + 64, 0, 1);
      }
      _Float16* dst = &As[row * APAD + c4];
      dst[0] = (_Float16)f.x; dst[1] = (_Float16)f.y;
      dst[2] = (_Float16)f.z; dst[3] = (_Float16)f.w;
    }
    // ---- cooperative W tile load: 64x64 f32, stored transposed [col][k]
#pragma unroll
    for (int q = 0; q < 4; ++q) {
      int idx = q * 256 + tid;           // 0..1023 float4 slots
      int kr  = idx >> 4;                // 0..63
      int c4  = (idx & 15) << 2;         // 0..60
      int c   = colBase + c4;
      float4 f = {0.f, 0.f, 0.f, 0.f};
      if (FULL || c + 3 < Ndim) {
        const float* src = W + (size_t)(kb + kr) * ldw + c;
        f = *(const float4*)src;
        if (q == 0 && kb + 64 < Kdim) __builtin_prefetch(src + (size_t)64 * ldw, 0, 1);
      }
      Bs[(c4 + 0) * BPAD + kr] = (_Float16)f.x;
      Bs[(c4 + 1) * BPAD + kr] = (_Float16)f.y;
      Bs[(c4 + 2) * BPAD + kr] = (_Float16)f.z;
      Bs[(c4 + 3) * BPAD + kr] = (_Float16)f.w;
    }
    __syncthreads();

    const _Float16* arow = &As[(wave * 16 + l15) * APAD];
#pragma unroll
    for (int s = 0; s < 2; ++s) {        // two 32-wide k sub-steps
      // ---- A fragment: 16x32 f16, contiguous k = 16*khalf + e (same as B)
      union { v16h v; _Float16 e[16]; } af;
#pragma unroll
      for (int v = 0; v < 8; ++v) {
        int k0 = s * 32 + khalf * 16 + v * 2;
        af.e[2 * v]     = arow[k0];
        af.e[2 * v + 1] = arow[k0 + 1];
      }
      // ---- 4 WMMAs along N
#pragma unroll
      for (int t = 0; t < 4; ++t) {
        union { v16h v; _Float16 e[16]; } bf;
        const _Float16* bcol = &Bs[(t * 16 + l15) * BPAD + s * 32 + khalf * 16];
#pragma unroll
        for (int v = 0; v < 16; ++v) bf.e[v] = bcol[v];
        acc[t] = __builtin_amdgcn_wmma_f32_16x16x32_f16(
            false, af.v, false, bf.v, (short)0, acc[t], false, false);
      }
    }
    __syncthreads();
  }

  // ---- epilogue: C/D layout lane=(col), VGPR v -> row v + 8*khalf
#pragma unroll
  for (int t = 0; t < 4; ++t) {
    union { v8f v; float e[8]; } u; u.v = acc[t];
#pragma unroll
    for (int v = 0; v < 8; ++v) {
      int r = rowBase + wave * 16 + khalf * 8 + v;
      int c = colBase + t * 16 + l15;
      if (FULL || (r < Mdim && c < Ndim)) {
        float val = u.e[v];
        if (bias) val += bias[c];
        if (act == 1) {  // tanh-approx GELU
          float x = val;
          val = 0.5f * x * (1.0f + tanhf(0.7978845608f * (x + 0.044715f * x * x * x)));
        }
        size_t o = (size_t)r * ldc + c;
        if (resid) val += resid[o];
        C[o] = val;
      }
    }
  }
}

// =====================================================================
// Patch + circular conv + sinusoidal PE  ->  H0[:, :LEN_S, :]
// =====================================================================
__global__ __launch_bounds__(256) void k_build_enc(
    const float* __restrict__ xe, const float* __restrict__ cw, float* __restrict__ H0)
{
  __shared__ float patch[144];   // [c*3 + k], c = p*3+m
  int s = blockIdx.x, b = blockIdx.y, tid = threadIdx.x;
  if (tid < 144) {
    int c = tid / 3, k = tid % 3;
    int p = c / 3,  m = c % 3;
    int srow = (s + k - 1 + LEN_S) & (LEN_S - 1);   // circular pad
    int j = srow * 8 + p; if (j > LLEN - 1) j = LLEN - 1;  // repeat-pad tail
    patch[tid] = xe[(size_t)(b * LLEN + j) * MFEAT + m];
  }
  __syncthreads();
#pragma unroll
  for (int q = 0; q < 3; ++q) {
    int d = tid + q * 256;
    const float* w = cw + (size_t)d * 144;
    float acc = 0.f;
    for (int i = 0; i < 144; ++i) acc += w[i] * patch[i];
    int jj = d >> 1;
    float ang = (float)s * expf(-(float)(2 * jj) * 0.011992631f);  // ln(1e4)/768
    float pe = (d & 1) ? cosf(ang) : sinf(ang);
    H0[((size_t)(b * TT + s)) * DD + d] = acc + pe;
  }
}

__global__ __launch_bounds__(256) void k_fill_prompt(
    const float* __restrict__ pr, float* __restrict__ H0)
{
  int idx = blockIdx.x * 256 + threadIdx.x;        // over BB*LEN_P*DD
  int b = idx / (LEN_P * DD);
  int r = idx % (LEN_P * DD);
  H0[((size_t)b * TT + LEN_S) * DD + r] = pr[r];
}

__global__ __launch_bounds__(256) void k_add_wpe(
    const float* __restrict__ H0, const float* __restrict__ wpe, float* __restrict__ H)
{
  int idx = blockIdx.x * 256 + threadIdx.x;        // over NTOK*DD
  int d = idx % DD;
  int t = (idx / DD) % TT;
  H[idx] = H0[idx] + wpe[t * DD + d];
}

// =====================================================================
// Edge cosine similarities + per-batch 2-segment softmax -> ew[B*E0]
// chain graph: edge 2i = (i -> i+1), edge 2i+1 = (i+1 -> i)
// =====================================================================
__global__ __launch_bounds__(256) void k_edge(
    const float* __restrict__ H0, float* __restrict__ EWp)
{
  __shared__ float nrm[TT];
  __shared__ float cs[TT - 1];
  int b = blockIdx.x, tid = threadIdx.x;
  if (tid < TT) {
    const float* x = H0 + ((size_t)(b * TT + tid)) * DD;
    float s = 0.f;
    for (int d = 0; d < DD; ++d) { float v = x[d]; s += v * v; }
    nrm[tid] = sqrtf(s);
  }
  __syncthreads();
  if (tid < TT - 1) {
    const float* x = H0 + ((size_t)(b * TT + tid)) * DD;
    const float* y = x + DD;
    float s = 0.f;
    for (int d = 0; d < DD; ++d) s += x[d] * y[d];
    cs[tid] = s / (nrm[tid] * nrm[tid + 1] + 1e-8f);
  }
  __syncthreads();
  if (tid < 2) {
    // tid==1 -> segment m=1 (S-S edges, i in [0,126]); tid==0 -> m=0 (i in [127,138])
    int lo = tid ? 0 : (LEN_S - 1);
    int hi = tid ? (LEN_S - 1) : (TT - 1);
    float mx = -3.0e38f;
    for (int i = lo; i < hi; ++i) mx = fmaxf(mx, cs[i]);
    float sum = 0.f;
    for (int i = lo; i < hi; ++i) sum += 2.f * expf(cs[i] - mx);  // both directions
    float denom = sum + 1e-8f;
    for (int i = lo; i < hi; ++i) {
      float w = expf(cs[i] - mx) / denom;
      EWp[b * E0 + 2 * i]     = w;
      EWp[b * E0 + 2 * i + 1] = w;
    }
  }
}

// =====================================================================
// LayerNorm over DD (act==1 adds exact GELU for the final stage)
// =====================================================================
__global__ __launch_bounds__(256) void k_layernorm(
    const float* __restrict__ X, const float* __restrict__ g, const float* __restrict__ bta,
    float* __restrict__ Y, int act)
{
  __shared__ float s1[256], s2[256];
  int row = blockIdx.x, tid = threadIdx.x;
  const float* x = X + (size_t)row * DD;
  float a0 = x[tid], a1 = x[tid + 256], a2 = x[tid + 512];
  s1[tid] = a0 + a1 + a2;
  s2[tid] = a0 * a0 + a1 * a1 + a2 * a2;
  __syncthreads();
  for (int off = 128; off > 0; off >>= 1) {
    if (tid < off) { s1[tid] += s1[tid + off]; s2[tid] += s2[tid + off]; }
    __syncthreads();
  }
  float mu   = s1[0] * (1.0f / DD);
  float var  = s2[0] * (1.0f / DD) - mu * mu;
  float rstd = rsqrtf(var + 1e-5f);
  float* y = Y + (size_t)row * DD;
#pragma unroll
  for (int q = 0; q < 3; ++q) {
    int i = tid + q * 256;
    float a = (q == 0) ? a0 : ((q == 1) ? a1 : a2);
    float v = (a - mu) * rstd * g[i] + bta[i];
    if (act == 1) v = 0.5f * v * (1.0f + erff(v * 0.70710678f));  // exact GELU
    y[i] = v;
  }
}

// =====================================================================
// Causal attention, one block per (batch, head); online softmax over
// 64-key LDS tiles.  qkv layout: row b*T+t, cols [q | k | v] of 768 each.
// =====================================================================
__global__ __launch_bounds__(256) void k_attention(
    const float* __restrict__ qkv, float* __restrict__ O)
{
  __shared__ float Ks[64][65];
  __shared__ float Vs[64][65];
  int bh = blockIdx.x;
  int b = bh / NHEAD, h = bh % NHEAD;
  int tid = threadIdx.x;
  const size_t rs = 3 * DD;
  const float* base = qkv + (size_t)b * TT * rs + h * HD;

  float qv[HD], ov[HD];
  float m = -3.0e38f, lsum = 0.f;
  int tq = tid;
  if (tq < TT) {
#pragma unroll
    for (int d = 0; d < HD; ++d) { qv[d] = base[(size_t)tq * rs + d]; ov[d] = 0.f; }
  }
  for (int kt = 0; kt < 3; ++kt) {
#pragma unroll
    for (int q = 0; q < 16; ++q) {
      int idx = q * 256 + tid;
      int r = idx >> 6, c = idx & 63;
      int kr = kt * 64 + r;
      float kvv = 0.f, vvv = 0.f;
      if (kr < TT) {
        kvv = qkv[((size_t)(b * TT + kr)) * rs + DD + h * HD + c];
        vvv = qkv[((size_t)(b * TT + kr)) * rs + 2 * DD + h * HD + c];
      }
      Ks[r][c] = kvv; Vs[r][c] = vvv;
    }
    __syncthreads();
    if (tq < TT) {
      int jmax = tq - kt * 64; if (jmax > 63) jmax = 63;
      for (int j = 0; j <= jmax; ++j) {
        float s = 0.f;
#pragma unroll
        for (int d = 0; d < HD; ++d) s += qv[d] * Ks[j][d];
        s *= 0.125f;                        // 1/sqrt(HD)
        float mn = fmaxf(m, s);
        float corr = expf(m - mn);
        float w = expf(s - mn);
        lsum = lsum * corr + w;
#pragma unroll
        for (int d = 0; d < HD; ++d) ov[d] = ov[d] * corr + w * Vs[j][d];
        m = mn;
      }
    }
    __syncthreads();
  }
  if (tq < TT) {
    float inv = 1.f / lsum;
    float* o = O + ((size_t)(b * TT + tq)) * DD + h * HD;
#pragma unroll
    for (int d = 0; d < HD; ++d) o[d] = ov[d] * inv;
  }
}

// ==================== graph stencil: out = h + ew-weighted neighbors ====
__global__ __launch_bounds__(256) void k_graph(
    const float* __restrict__ H, const float* __restrict__ EWp, float* __restrict__ Out)
{
  int idx = blockIdx.x * 256 + threadIdx.x;       // over NTOK*DD
  int t = (idx / DD) % TT;
  int b = idx / (DD * TT);
  float v = H[idx];
  if (t > 0)      v += EWp[b * E0 + 2 * (t - 1)] * H[idx - DD];
  if (t < TT - 1) v += EWp[b * E0 + 2 * t + 1]  * H[idx + DD];
  Out[idx] = v;
}

// ==================== l_h fold / l2s projection / l2s add ===============
__global__ __launch_bounds__(256) void k_lh_fold(float* __restrict__ LH)
{
  int idx = blockIdx.x * 256 + threadIdx.x;       // over BB*DD
  int b = idx / DD, d = idx % DD;
  size_t o = (size_t)b * 2 * DD + d;
  float s = LH[o] + LH[o + DD];
  LH[o] = s; LH[o + DD] = s;
}

__global__ __launch_bounds__(256) void k_lproj(
    const float* __restrict__ LH, const float* __restrict__ w, const float* __restrict__ bta,
    float* __restrict__ P)
{
  int idx = blockIdx.x * 256 + threadIdx.x;       // over BB*2*DD
  int d = idx % DD;
  int r = (idx / DD) & 1;
  int b = idx / (2 * DD);
  const float* lh = LH + (size_t)b * 2 * DD + r * DD;
  float acc = bta[d];
  for (int k = 0; k < 128; ++k) acc += lh[k] * w[k * DD + d];
  P[idx] = acc;
}

__global__ __launch_bounds__(256) void k_l2s_add(
    const float* __restrict__ P, const float* __restrict__ wptr, float* __restrict__ H)
{
  int idx = blockIdx.x * 256 + threadIdx.x;       // over NTOK*DD
  int d = idx % DD;
  int t = (idx / DD) % TT;
  int b = idx / (DD * TT);
  float wl = *wptr;
  float coef = (t < LEN_S) ? (wl / (float)LEN_S) : (wl / (float)LEN_P);
  int row = (t < LEN_S) ? 0 : 1;
  H[idx] += coef * P[(size_t)b * 2 * DD + row * DD + d];
}

// ==================== final batch-wide LN stats + output ================
__global__ __launch_bounds__(256) void k_batch_stats(
    const float* __restrict__ Z, float* __restrict__ stats)
{
  __shared__ float s1[256], s2[256];
  int b = blockIdx.x, tid = threadIdx.x;
  const float* z = Z + (size_t)b * ZN;
  float sum = 0.f, sq = 0.f;
  for (int i = tid; i < ZN; i += 256) { float v = z[i]; sum += v; sq += v * v; }
  s1[tid] = sum; s2[tid] = sq;
  __syncthreads();
  for (int off = 128; off > 0; off >>= 1) {
    if (tid < off) { s1[tid] += s1[tid + off]; s2[tid] += s2[tid + off]; }
    __syncthreads();
  }
  if (tid == 0) {
    float mu = s1[0] / (float)ZN;
    float var = s2[0] / (float)ZN - mu * mu;
    stats[b * 2] = mu;
    stats[b * 2 + 1] = rsqrtf(var + 1e-5f);
  }
}

__global__ __launch_bounds__(256) void k_out(
    const float* __restrict__ Z, const float* __restrict__ stats,
    const float* __restrict__ g, const float* __restrict__ bta,
    const float* __restrict__ Wout, const float* __restrict__ bout,
    float* __restrict__ out)
{
  __shared__ float red[256][10];
  int b = blockIdx.x, tid = threadIdx.x;
  float mu = stats[b * 2], rstd = stats[b * 2 + 1];
  const float* z = Z + (size_t)b * ZN;
  float acc[10];
#pragma unroll
  for (int n = 0; n < 10; ++n) acc[n] = 0.f;
  for (int k = tid; k < ZN; k += 256) {
    float v = (z[k] - mu) * rstd * g[k] + bta[k];
#pragma unroll
    for (int n = 0; n < 10; ++n) acc[n] += v * Wout[(size_t)k * 10 + n];
  }
#pragma unroll
  for (int n = 0; n < 10; ++n) red[tid][n] = acc[n];
  __syncthreads();
  for (int off = 128; off > 0; off >>= 1) {
    if (tid < off) {
#pragma unroll
      for (int n = 0; n < 10; ++n) red[tid][n] += red[tid + off][n];
    }
    __syncthreads();
  }
  if (tid < 10) out[b * 10 + tid] = red[0][tid] + bout[tid];
}

// =====================================================================
// Host launcher
// =====================================================================
extern "C" void kernel_launch(void* const* d_in, const int* in_sizes, int n_in,
                              void* d_out, int out_size, void* d_ws, size_t ws_size,
                              hipStream_t stream)
{
  (void)in_sizes; (void)n_in; (void)out_size; (void)ws_size;
  const float* x_enc   = (const float*)d_in[0];
  const float* conv_w  = (const float*)d_in[2];
  const float* prompt  = (const float*)d_in[3];
  const float* wpe     = (const float*)d_in[4];
  const float* ln1_g   = (const float*)d_in[5];
  const float* ln1_b   = (const float*)d_in[6];
  const float* attn_w  = (const float*)d_in[7];
  const float* attn_b  = (const float*)d_in[8];
  const float* proj_w  = (const float*)d_in[9];
  const float* proj_b  = (const float*)d_in[10];
  const float* ln2_g   = (const float*)d_in[11];
  const float* ln2_b   = (const float*)d_in[12];
  const float* fc1_w   = (const float*)d_in[13];
  const float* fc1_b   = (const float*)d_in[14];
  const float* fc2_w   = (const float*)d_in[15];
  const float* fc2_b   = (const float*)d_in[16];
  const float* lnf_g   = (const float*)d_in[17];
  const float* lnf_b   = (const float*)d_in[18];
  const float* ll1_w   = (const float*)d_in[19];
  const float* ll1_b   = (const float*)d_in[20];
  const float* ll0_w   = (const float*)d_in[21];
  const float* ll0_b   = (const float*)d_in[22];
  const float* l2s_w   = (const float*)d_in[23];
  const float* l2s_b   = (const float*)d_in[24];
  const float* w_l2s   = (const float*)d_in[25];
  const float* lnp_g   = (const float*)d_in[26];
  const float* lnp_b   = (const float*)d_in[27];
  const float* out_w   = (const float*)d_in[28];
  const float* out_b   = (const float*)d_in[29];

  // ---- workspace carving (256B aligned) ----
  char* p = (char*)d_ws;
  auto carve = [&](size_t nfloat) -> float* {
    float* r = (float*)p;
    p += ((nfloat * sizeof(float)) + 255) & ~(size_t)255;
    return r;
  };
  float* H0    = carve((size_t)NTOK * DD);
  float* HA    = carve((size_t)NTOK * DD);
  float* HB    = carve((size_t)NTOK * DD);
  float* BIG   = carve((size_t)NTOK * DMLP);
  float* LH    = carve((size_t)BB * 2 * DD);
  float* LPROJ = carve((size_t)BB * 2 * DD);
  float* EW    = carve((size_t)BB * E0);
  float* STATS = carve((size_t)BB * 2);

  dim3 blk(256);
  auto gemm = [&](const float* A, int lda, const float* W, int ldw,
                  const float* bias, const float* resid,
                  float* C, int ldc, int Md, int Nd, int Kd, int act) {
    dim3 g((Nd + 63) / 64, (Md + 127) / 128);
    bool full = (Md % 128 == 0) && (Nd % 64 == 0);
    if (full)
      k_gemm<true><<<g, blk, 0, stream>>>(A, lda, W, ldw, bias, resid, C, ldc, Md, Nd, Kd, act);
    else
      k_gemm<false><<<g, blk, 0, stream>>>(A, lda, W, ldw, bias, resid, C, ldc, Md, Nd, Kd, act);
  };

  // ---- front end ----
  k_build_enc<<<dim3(LEN_S, BB), blk, 0, stream>>>(x_enc, conv_w, H0);
  k_fill_prompt<<<(BB * LEN_P * DD) / 256, blk, 0, stream>>>(prompt, H0);
  k_edge<<<BB, blk, 0, stream>>>(H0, EW);

  // l_h init: o1 @ ll1_w and o2 @ ll0_w (strided views into H0 / LH)
  gemm(H0,                TT * DD, ll1_w, DD, ll1_b, nullptr, LH,      2 * DD,
       BB, DD, LEN_S * DD, 0);
  gemm(H0 + LEN_S * DD,   TT * DD, ll0_w, DD, ll0_b, nullptr, LH + DD, 2 * DD,
       BB, DD, LEN_P * DD, 0);

  k_add_wpe<<<(NTOK * DD) / 256, blk, 0, stream>>>(H0, wpe, HA);

  float* hcur = HA;
  float* haux = HB;
  for (int l = 0; l < NLAY; ++l) {
    const float* aw  = attn_w + (size_t)l * DD * 3 * DD;
    const float* ab  = attn_b + (size_t)l * 3 * DD;
    const float* pw  = proj_w + (size_t)l * DD * DD;
    const float* pb  = proj_b + (size_t)l * DD;
    const float* f1w = fc1_w  + (size_t)l * DD * DMLP;
    const float* f1b = fc1_b  + (size_t)l * DMLP;
    const float* f2w = fc2_w  + (size_t)l * DMLP * DD;
    const float* f2b = fc2_b  + (size_t)l * DD;

    k_layernorm<<<NTOK, blk, 0, stream>>>(hcur, ln1_g + l * DD, ln1_b + l * DD, haux, 0);
    gemm(haux, DD, aw, 3 * DD, ab, nullptr, BIG, 3 * DD, NTOK, 3 * DD, DD, 0);
    k_attention<<<BB * NHEAD, blk, 0, stream>>>(BIG, haux);
    gemm(haux, DD, pw, DD, pb, hcur, hcur, DD, NTOK, DD, DD, 0);   // in-place resid
    k_graph<<<(NTOK * DD) / 256, blk, 0, stream>>>(hcur, EW, haux);
    { float* t = hcur; hcur = haux; haux = t; }                    // haux = new h
    k_layernorm<<<NTOK, blk, 0, stream>>>(hcur, ln2_g + l * DD, ln2_b + l * DD, haux, 0);
    gemm(haux, DD, f1w, DMLP, f1b, nullptr, BIG, DMLP, NTOK, DMLP, DD, 1);  // gelu
    gemm(BIG, DMLP, f2w, DD, f2b, hcur, hcur, DD, NTOK, DD, DMLP, 0);       // resid
    k_lh_fold<<<(BB * DD) / 256, blk, 0, stream>>>(LH);
    k_lproj<<<(BB * 2 * DD) / 256, blk, 0, stream>>>(LH, l2s_w, l2s_b, LPROJ);
    k_l2s_add<<<(NTOK * DD) / 256, blk, 0, stream>>>(LPROJ, w_l2s, hcur);
  }

  // ---- final: LN + exact GELU -> batch-wide LN -> 10-class head ----
  k_layernorm<<<NTOK, blk, 0, stream>>>(hcur, lnf_g, lnf_b, haux, 1);
  k_batch_stats<<<BB, blk, 0, stream>>>(haux, STATS);
  k_out<<<BB, blk, 0, stream>>>(haux, STATS, lnp_g, lnp_b, out_w, out_b, (float*)d_out);
}
